// Model_55138790146624
// MI455X (gfx1250) — matrix operand order repaired and checked by reading, not compiled, not run on hardware
//
#include <hip/hip_runtime.h>
#include <hip/hip_bf16.h>
#include <cstdint>

// Model dims
#define VOCABN 128
#define HDIM   1024
#define BATCHN 64
#define SEQN   512
#define G4H    4096   // 4*HDIM gate width

// ---------------- CDNA5 WMMA plumbing (gfx1250, wave32) ----------------
typedef __attribute__((ext_vector_type(16))) __bf16 v16bf;
typedef __attribute__((ext_vector_type(8)))  float  v8f;

union FragU { uint4 q[2]; v16bf v; };

__device__ __forceinline__ unsigned short f32_to_bf16(float f) {
  unsigned int u = __float_as_uint(f);
  u += 0x7FFFu + ((u >> 16) & 1u);   // round-to-nearest-even
  return (unsigned short)(u >> 16);
}

// A-matrix fragment (16x32, MxK, bf16). Lane l holds row m = l&15.
// Element i <-> k = (i/8)*16 + ((l>>4)&1)*8 + (i%8)  (per ISA 16-bit A layout)
__device__ __forceinline__ v16bf load_fragA(const unsigned short* __restrict__ row,
                                            int k0, int lane) {
  FragU u;
  const int kh = ((lane >> 4) & 1) << 3;
  u.q[0] = *(const uint4*)(row + k0 + kh);
  u.q[1] = *(const uint4*)(row + k0 + 16 + kh);
  return u.v;
}

// B-matrix fragment (32x16, KxN) sourced from a row-major (N,K) weight matrix,
// i.e. B[k][n] = W[n][k]. Lane l holds col n = l&15; element i <-> k = (i/8)*16 + ((l>>4)&1)*8 + (i%8).
__device__ __forceinline__ v16bf load_fragB(const unsigned short* __restrict__ row,
                                            int k0, int lane) {
  FragU u;
  const int kh = ((lane >> 4) & 1) << 3;
  u.q[0] = *(const uint4*)(row + k0 + kh);
  u.q[1] = *(const uint4*)(row + k0 + 16 + kh);
  return u.v;
}

__device__ __forceinline__ v8f wmma_bf16(v16bf a, v16bf b, v8f c) {
  // (neg_a, A, neg_b, B, c_mod, C, reuse_a, reuse_b)
  return __builtin_amdgcn_wmma_f32_16x16x32_bf16(false, a, false, b, (short)0, c,
                                                 false, false);
}

__device__ __forceinline__ float sigmoidf_(float x) {
  return 1.0f / (1.0f + __expf(-x));
}

// ---------------- LSTM gate GEMM (low register pressure variant) ------------
// acc[g] (+)= A_lds(16xK) x W(g*H + hcol .. +15, :)^T   for g = i,f,g,o
// A tile lives in LDS (staged cooperatively by the block); W streams from L2.
// acc[g] element v <-> D[m = 8*(lane>>4) + v][n = hcol + (lane&15)]
__device__ __forceinline__ void gate_gemm16(v8f acc[4],
                                            const unsigned short* Alds, // 16 x HDIM (LDS)
                                            const unsigned short* __restrict__ W,
                                            int hcol, int lane) {
  const int ln = lane & 15;
  const unsigned short* wrow[4];
#pragma unroll
  for (int g = 0; g < 4; ++g)
    wrow[g] = W + (size_t)(g * HDIM + hcol + ln) * HDIM;
  const unsigned short* arow = Alds + (size_t)ln * HDIM;

  for (int k0 = 0; k0 < HDIM; k0 += 32) {
    v16bf af = load_fragA(arow, k0, lane);     // ds_load_b128 x2
#pragma unroll
    for (int g = 0; g < 4; ++g)
      acc[g] = wmma_bf16(af, load_fragB(wrow[g], k0, lane), acc[g]);
  }
}

// Cooperative stage of a 16 x HDIM bf16 tile (32 KB) into LDS.
__device__ __forceinline__ void stage_tile(unsigned short* dstLds,
                                           const unsigned short* __restrict__ src,
                                           int tid) {
  uint4* d = (uint4*)dstLds;
  const uint4* s = (const uint4*)src;
#pragma unroll
  for (int i = 0; i < (16 * HDIM / 8) / 256; ++i)   // 8 iterations
    d[tid + i * 256] = s[tid + i * 256];
}

// ---------------- LSTM step kernels -----------------------------------------
// grid = 32 blocks (4 m-tiles x 8 h-groups) x 256 threads (8 waves).
// Each wave owns a 16(batch) x 16(hidden) tile across all four gates, so the
// whole cell update fuses in-register. 4 accumulators = 32 VGPRs: no spills.

__global__ void lstm_step0(const unsigned short* __restrict__ Whh,   // (4H,H) bf16
                           const unsigned short* __restrict__ hprev, // (B,H) bf16
                           const float* __restrict__ bias,           // (4H) b_ih0+b_hh0
                           const float* __restrict__ xsum,           // (4H) rowsum(Wih0)
                           const int* __restrict__ rowhit,           // (V) hit flags
                           int t,
                           float* __restrict__ cbuf,                 // (B,H) f32
                           unsigned short* __restrict__ hout) {      // (B,H) bf16
  __shared__ __align__(16) unsigned short sA[16 * HDIM];   // 32 KB
  const int tid   = threadIdx.x;
  const int lane  = tid & 31;
  const int wave  = tid >> 5;
  const int mtile = blockIdx.x >> 3;          // 0..3  (batch tile)
  const int hcol  = (blockIdx.x & 7) * 128 + wave * 16;

  stage_tile(sA, hprev + (size_t)mtile * 16 * HDIM, tid);
  __syncthreads();

  v8f vzero = {};
  v8f acc[4];
#pragma unroll
  for (int g = 0; g < 4; ++g) acc[g] = vzero;

  gate_gemm16(acc, sA, Whh, hcol, lane);

  const int ln  = lane & 15;
  const int mhi = ((lane >> 4) & 1) * 8;
  const int h   = hcol + ln;
  // one-hot "bug": only (b==0, t<128) rows can be non-zero; the contribution
  // is then the row-sum of Wih0, gated by whether token value t appears in x.
  const bool xact = (mtile == 0) && (t < VOCABN) && (rowhit[t] != 0);

#pragma unroll
  for (int v = 0; v < 8; ++v) {
    const int m = mtile * 16 + mhi + v;      // batch index
    float xi = 0.f, xf = 0.f, xg = 0.f, xo = 0.f;
    if (xact && m == 0) {
      xi = xsum[h];
      xf = xsum[HDIM + h];
      xg = xsum[2 * HDIM + h];
      xo = xsum[3 * HDIM + h];
    }
    const float gi = sigmoidf_(acc[0][v] + bias[h] + xi);
    const float gf = sigmoidf_(acc[1][v] + bias[HDIM + h] + xf);
    const float gg = tanhf    (acc[2][v] + bias[2 * HDIM + h] + xg);
    const float go = sigmoidf_(acc[3][v] + bias[3 * HDIM + h] + xo);
    const size_t idx = (size_t)m * HDIM + h;
    const float cn = gf * cbuf[idx] + gi * gg;
    cbuf[idx] = cn;
    hout[idx] = f32_to_bf16(go * tanhf(cn));
  }
}

__global__ void lstm_step1(const unsigned short* __restrict__ Wih,   // (4H,H) bf16
                           const unsigned short* __restrict__ Whh,   // (4H,H) bf16
                           const unsigned short* __restrict__ xin,   // layer0 h_t (B,H)
                           const unsigned short* __restrict__ hprev, // layer1 h_{t-1}
                           const float* __restrict__ bias,           // (4H)
                           float* __restrict__ cbuf,
                           unsigned short* __restrict__ hout) {
  __shared__ __align__(16) unsigned short sA[16 * HDIM];   // 32 KB, reused twice
  const int tid   = threadIdx.x;
  const int lane  = tid & 31;
  const int wave  = tid >> 5;
  const int mtile = blockIdx.x >> 3;
  const int hcol  = (blockIdx.x & 7) * 128 + wave * 16;

  v8f vzero = {};
  v8f acc[4];
#pragma unroll
  for (int g = 0; g < 4; ++g) acc[g] = vzero;

  // input GEMM (K=1024)
  stage_tile(sA, xin + (size_t)mtile * 16 * HDIM, tid);
  __syncthreads();
  gate_gemm16(acc, sA, Wih, hcol, lane);
  __syncthreads();

  // recurrent GEMM (K=1024)
  stage_tile(sA, hprev + (size_t)mtile * 16 * HDIM, tid);
  __syncthreads();
  gate_gemm16(acc, sA, Whh, hcol, lane);

  const int ln  = lane & 15;
  const int mhi = ((lane >> 4) & 1) * 8;
  const int h   = hcol + ln;

#pragma unroll
  for (int v = 0; v < 8; ++v) {
    const int m = mtile * 16 + mhi + v;
    const float gi = sigmoidf_(acc[0][v] + bias[h]);
    const float gf = sigmoidf_(acc[1][v] + bias[HDIM + h]);
    const float gg = tanhf    (acc[2][v] + bias[2 * HDIM + h]);
    const float go = sigmoidf_(acc[3][v] + bias[3 * HDIM + h]);
    const size_t idx = (size_t)m * HDIM + h;
    const float cn = gf * cbuf[idx] + gi * gg;
    cbuf[idx] = cn;
    hout[idx] = f32_to_bf16(go * tanhf(cn));
  }
}

// ---------------- Classifier GEMMs ------------------------------------------
// Each wave computes a 64(M) x 16(N) tile, K=1024. 4 accumulators: no spills.

__global__ void fc_relu_bf16(const unsigned short* __restrict__ A,  // (M,1024) bf16
                             const unsigned short* __restrict__ W,  // (1024,1024) bf16
                             const float* __restrict__ bias,        // (1024) f32
                             unsigned short* __restrict__ out) {    // (M,1024) bf16
  const int lane  = threadIdx.x & 31;
  const int wv    = blockIdx.x * 8 + (threadIdx.x >> 5);
  const int Ntile = HDIM / 16;                 // 64
  const int mtile = wv / Ntile;
  const int ntile = wv % Ntile;
  const int ln    = lane & 15;

  const unsigned short* wrow = W + (size_t)(ntile * 16 + ln) * HDIM;
  const unsigned short* arow[4];
#pragma unroll
  for (int mt = 0; mt < 4; ++mt)
    arow[mt] = A + (size_t)(mtile * 64 + mt * 16 + ln) * HDIM;

  v8f vzero = {};
  v8f acc[4];
#pragma unroll
  for (int mt = 0; mt < 4; ++mt) acc[mt] = vzero;

  for (int k0 = 0; k0 < HDIM; k0 += 32) {
    v16bf bf = load_fragB(wrow, k0, lane);
#pragma unroll
    for (int mt = 0; mt < 4; ++mt)
      acc[mt] = wmma_bf16(load_fragA(arow[mt], k0, lane), bf, acc[mt]);
  }

  const int mhi = ((lane >> 4) & 1) * 8;
  const int n   = ntile * 16 + ln;
  const float bn = bias[n];
#pragma unroll
  for (int mt = 0; mt < 4; ++mt) {
#pragma unroll
    for (int v = 0; v < 8; ++v) {
      const int m = mtile * 64 + mt * 16 + mhi + v;
      float val = acc[mt][v] + bn;
      val = val > 0.f ? val : 0.f;             // ReLU
      out[(size_t)m * HDIM + n] = f32_to_bf16(val);
    }
  }
}

__global__ void fc_logits(const unsigned short* __restrict__ A,  // (M,1024) bf16
                          const unsigned short* __restrict__ W,  // (128,1024) bf16
                          const float* __restrict__ bias,        // (128) f32
                          float* __restrict__ out) {             // (M,128) f32
  const int lane  = threadIdx.x & 31;
  const int wv    = blockIdx.x * 8 + (threadIdx.x >> 5);
  const int Ntile = VOCABN / 16;               // 8
  const int mtile = wv / Ntile;
  const int ntile = wv % Ntile;
  const int ln    = lane & 15;

  const unsigned short* wrow = W + (size_t)(ntile * 16 + ln) * HDIM;
  const unsigned short* arow[4];
#pragma unroll
  for (int mt = 0; mt < 4; ++mt)
    arow[mt] = A + (size_t)(mtile * 64 + mt * 16 + ln) * HDIM;

  v8f vzero = {};
  v8f acc[4];
#pragma unroll
  for (int mt = 0; mt < 4; ++mt) acc[mt] = vzero;

  for (int k0 = 0; k0 < HDIM; k0 += 32) {
    v16bf bf = load_fragB(wrow, k0, lane);
#pragma unroll
    for (int mt = 0; mt < 4; ++mt)
      acc[mt] = wmma_bf16(load_fragA(arow[mt], k0, lane), bf, acc[mt]);
  }

  const int mhi = ((lane >> 4) & 1) * 8;
  const int n   = ntile * 16 + ln;
  const float bn = bias[n];
#pragma unroll
  for (int mt = 0; mt < 4; ++mt) {
#pragma unroll
    for (int v = 0; v < 8; ++v) {
      const int m = mtile * 64 + mt * 16 + mhi + v;
      out[(size_t)m * VOCABN + n] = acc[mt][v] + bn;
    }
  }
}

// ---------------- Prep kernels ----------------------------------------------

__global__ void cvt_bf16_kernel(const float* __restrict__ src,
                                unsigned short* __restrict__ dst, int n) {
  const int i = blockIdx.x * blockDim.x + threadIdx.x;
  if (i < n) dst[i] = f32_to_bf16(src[i]);
}

__global__ void bias_sum_kernel(const float* __restrict__ a,
                                const float* __restrict__ b,
                                float* __restrict__ o, int n) {
  const int i = blockIdx.x * blockDim.x + threadIdx.x;
  if (i < n) o[i] = a[i] + b[i];
}

__global__ void rowsum_kernel(const float* __restrict__ W,   // (4H, V) f32
                              float* __restrict__ o) {       // (4H) f32
  const int n = blockIdx.x * blockDim.x + threadIdx.x;
  if (n < G4H) {
    float s = 0.f;
#pragma unroll 8
    for (int k = 0; k < VOCABN; ++k) s += W[(size_t)n * VOCABN + k];
    o[n] = s;
  }
}

__global__ void rowhit_kernel(const int* __restrict__ x, int* __restrict__ hit,
                              int n) {
  const int i = blockIdx.x * blockDim.x + threadIdx.x;
  if (i < n) {
    const int tok = x[i];
    if (tok >= 0 && tok < VOCABN) atomicOr(&hit[tok], 1);
  }
}

// ---------------- Host orchestration ----------------------------------------

extern "C" void kernel_launch(void* const* d_in, const int* in_sizes, int n_in,
                              void* d_out, int out_size, void* d_ws, size_t ws_size,
                              hipStream_t stream) {
  (void)in_sizes; (void)n_in; (void)out_size; (void)ws_size;

  const int*   x     = (const int*)  d_in[0];
  const float* Wih0  = (const float*)d_in[1];
  const float* Whh0  = (const float*)d_in[2];
  const float* bih0  = (const float*)d_in[3];
  const float* bhh0  = (const float*)d_in[4];
  const float* Wih1  = (const float*)d_in[5];
  const float* Whh1  = (const float*)d_in[6];
  const float* bih1  = (const float*)d_in[7];
  const float* bhh1  = (const float*)d_in[8];
  const float* W1    = (const float*)d_in[9];
  const float* b1    = (const float*)d_in[10];
  const float* W2    = (const float*)d_in[11];
  const float* b2    = (const float*)d_in[12];
  float*       out   = (float*)d_out;

  // Workspace carve-up (~225 MB total)
  char* p = (char*)d_ws;
  auto alloc = [&](size_t bytes) -> void* {
    void* r = (void*)p;
    p += (bytes + 255) & ~(size_t)255;
    return r;
  };
  unsigned short* Whh0b  = (unsigned short*)alloc((size_t)G4H * HDIM * 2);
  unsigned short* Wih1b  = (unsigned short*)alloc((size_t)G4H * HDIM * 2);
  unsigned short* Whh1b  = (unsigned short*)alloc((size_t)G4H * HDIM * 2);
  unsigned short* W1b    = (unsigned short*)alloc((size_t)HDIM * HDIM * 2);
  unsigned short* W2b    = (unsigned short*)alloc((size_t)VOCABN * HDIM * 2);
  float*          bias0  = (float*)alloc((size_t)G4H * 4);
  float*          bias1  = (float*)alloc((size_t)G4H * 4);
  float*          xsum0  = (float*)alloc((size_t)G4H * 4);
  int*            rowhit = (int*)  alloc((size_t)VOCABN * 4);
  float*          c0     = (float*)alloc((size_t)BATCHN * HDIM * 4);
  float*          c1     = (float*)alloc((size_t)BATCHN * HDIM * 4);
  // (T+1) slots: slot 0 is the zero initial state; slot t+1 holds h_t.
  unsigned short* h0_all = (unsigned short*)alloc((size_t)(SEQN + 1) * BATCHN * HDIM * 2);
  unsigned short* h1_all = (unsigned short*)alloc((size_t)(SEQN + 1) * BATCHN * HDIM * 2);
  unsigned short* hmid   = (unsigned short*)alloc((size_t)SEQN * BATCHN * HDIM * 2);

  // Zero-init per-call state (bf16 zero == 0x0000)
  hipMemsetAsync(rowhit, 0, (size_t)VOCABN * 4, stream);
  hipMemsetAsync(c0,     0, (size_t)BATCHN * HDIM * 4, stream);
  hipMemsetAsync(c1,     0, (size_t)BATCHN * HDIM * 4, stream);
  hipMemsetAsync(h0_all, 0, (size_t)BATCHN * HDIM * 2, stream);
  hipMemsetAsync(h1_all, 0, (size_t)BATCHN * HDIM * 2, stream);

  // Weight conversion to bf16 (stays L2-resident: ~27.5 MB << 192 MB L2)
  const int nW = G4H * HDIM;
  cvt_bf16_kernel<<<(nW + 255) / 256, 256, 0, stream>>>(Whh0, Whh0b, nW);
  cvt_bf16_kernel<<<(nW + 255) / 256, 256, 0, stream>>>(Wih1, Wih1b, nW);
  cvt_bf16_kernel<<<(nW + 255) / 256, 256, 0, stream>>>(Whh1, Whh1b, nW);
  cvt_bf16_kernel<<<(HDIM * HDIM + 255) / 256, 256, 0, stream>>>(W1, W1b, HDIM * HDIM);
  cvt_bf16_kernel<<<(VOCABN * HDIM + 255) / 256, 256, 0, stream>>>(W2, W2b, VOCABN * HDIM);

  bias_sum_kernel<<<(G4H + 255) / 256, 256, 0, stream>>>(bih0, bhh0, bias0, G4H);
  bias_sum_kernel<<<(G4H + 255) / 256, 256, 0, stream>>>(bih1, bhh1, bias1, G4H);
  rowsum_kernel<<<(G4H + 255) / 256, 256, 0, stream>>>(Wih0, xsum0);
  rowhit_kernel<<<(BATCHN * SEQN + 255) / 256, 256, 0, stream>>>(x, rowhit, BATCHN * SEQN);

  // Recurrence: 512 steps, two fused WMMA+cell kernels per step.
  const size_t hstep = (size_t)BATCHN * HDIM;
  for (int t = 0; t < SEQN; ++t) {
    lstm_step0<<<32, 256, 0, stream>>>(Whh0b,
                                       h0_all + (size_t)t * hstep,
                                       bias0, xsum0, rowhit, t, c0,
                                       h0_all + (size_t)(t + 1) * hstep);
    lstm_step1<<<32, 256, 0, stream>>>(Wih1b, Whh1b,
                                       h0_all + (size_t)(t + 1) * hstep,
                                       h1_all + (size_t)t * hstep,
                                       bias1, c1,
                                       h1_all + (size_t)(t + 1) * hstep);
  }

  // Classifier over all (T*B, H) rows; h1_all slots 1..T are exactly that.
  const int Mrows = SEQN * BATCHN;                       // 32768
  fc_relu_bf16<<<(Mrows / 64) * (HDIM / 16) / 8, 256, 0, stream>>>(
      h1_all + hstep, W1b, b1, hmid);
  fc_logits<<<(Mrows / 64) * (VOCABN / 16) / 8, 256, 0, stream>>>(
      hmid, W2b, b2, out);
}
